// DGCNN_56624848831033
// MI455X (gfx1250) — matrix operand order repaired and checked by reading, not compiled, not run on hardware
//
#include <hip/hip_runtime.h>

// ---------------------------------------------------------------------------
// DGCNN forward for MI455X (gfx1250, wave32, WMMA, async-to-LDS).
// All GEMMs: v_wmma_f32_16x16x32_bf16 (bf16 operands, f32 accumulate).
// EdgeConv factored: h = y1[j] + (y2[i] - y1[i]) so the K=20 neighbor loop is
// a gather+max epilogue, not a GEMM. All K dims padded to 32 and rows 32B
// aligned so the GEMM inner loop has no guards; staging is double-buffered
// GLOBAL_LOAD_ASYNC_TO_LDS_B128 when the builtin exists.
// ---------------------------------------------------------------------------

typedef __attribute__((ext_vector_type(16))) __bf16 v16bf;
typedef __attribute__((ext_vector_type(8)))  __bf16 v8bf;
typedef __attribute__((ext_vector_type(8)))  float  v8f;
typedef __attribute__((ext_vector_type(4)))  int    i4;

#define NEGINF (-3.0e38f)
#define BM 128
#define BN 64
#define BK 32
#define LDSS 40   // 32 + 8 pad (80B rows -> 16B aligned, bank friendly)

#if defined(__HIP_DEVICE_COMPILE__) && __has_builtin(__builtin_amdgcn_global_load_async_to_lds_b128)
#define HAVE_ASYNC 1
#else
#define HAVE_ASYNC 0
#endif

#if HAVE_ASYNC
typedef __attribute__((address_space(1))) i4 gi4;   // global int4
typedef __attribute__((address_space(3))) i4 li4;   // LDS int4
static __device__ __forceinline__ void async_cp16(void* lds, const void* g){
  // copies 16B (8 bf16) global -> LDS, tracked by ASYNCcnt
  __builtin_amdgcn_global_load_async_to_lds_b128((gi4*)g, (li4*)lds, 0, 0);
}
static __device__ __forceinline__ void wait_async_all(){
#if __has_builtin(__builtin_amdgcn_s_wait_asynccnt)
  __builtin_amdgcn_s_wait_asynccnt(0);
#else
  asm volatile("s_wait_asynccnt 0x0" ::: "memory");
#endif
}
#endif

static __device__ __forceinline__ float lrelu(float h){ return h >= 0.f ? h : 0.2f*h; }

// ---------------- prep kernels ----------------
// pack weight slice [O, C] (row stride ldsrc, col offset off) -> bf16 [O, Kpad], zero-pad
__global__ void k_pack_w(const float* __restrict__ src, __bf16* __restrict__ dst,
                         int O, int C, int off, int ldsrc, int Kpad){
  int i = blockIdx.x*256 + threadIdx.x;
  if (i >= O*Kpad) return;
  int o = i / Kpad, k = i - o*Kpad;
  dst[i] = (k < C) ? (__bf16)src[(size_t)o*ldsrc + off + k] : (__bf16)0.f;
}

// x: [B][6][N] f32 (channel-major) -> X0: [B][N][32] bf16 (point-major, zero pad)
__global__ void k_prep_x0(const float* __restrict__ x, __bf16* __restrict__ X0, int N){
  int n = blockIdx.x*256 + threadIdx.x;
  int b = blockIdx.y;
  if (n >= N) return;
  const float* xb = x + (size_t)b*6*N;
  __bf16* d = X0 + ((size_t)b*N + n)*32;
#pragma unroll
  for (int c=0;c<6;++c) d[c] = (__bf16)xb[(size_t)c*N + n];
#pragma unroll
  for (int c=6;c<32;++c) d[c] = (__bf16)0.f;
}

// per-point |x|^2 in f32 from bf16 activations
__global__ void k_sumsq(const __bf16* __restrict__ X, long long strX, int lda, int C,
                        float* __restrict__ sq, int N){
  int n = blockIdx.x*256 + threadIdx.x;
  int b = blockIdx.y;
  if (n >= N) return;
  const __bf16* p = X + (size_t)b*strX + (size_t)n*lda;
  float a = 0.f;
  for (int c=0;c<C;++c){ float v = (float)p[c]; a += v*v; }
  sq[(size_t)b*N + n] = a;
}

// ---------------- generic bf16 WMMA GEMM ----------------
// C[m, c] = sum_k A[m, k] * B[c, k]   (B row-major [Nc, K]: packed weights, or
// X itself for the Gram/distance matrix).
// Preconditions (guaranteed by layouts): M%128==0, Nc%64==0, Kc%32==0,
// A/B rows 32B aligned (lda/ldb multiples of 16 bf16, base 32B aligned).
struct GemmP {
  const __bf16* A; long long strA; int lda;
  const __bf16* B; long long strB; int ldb;
  int M, Nc, Kc;
  float*  outF; __bf16* outH; long long strO; int ldo;
  const float* s; const float* t;             // folded BN (EP==1)
  const float* sq;                            // |x|^2 per point (EP==2)
};

// 16-bit A 16x32 fragment: lane<16 holds K {0..7,16..23}, lane>=16 {8..15,24..31}
static __device__ __forceinline__ v16bf ldsA_frag(const __bf16* As, int row, int ksel){
  v8bf lo = *(const v8bf*)(As + row*LDSS + ksel);
  v8bf hi = *(const v8bf*)(As + row*LDSS + ksel + 16);
  return __builtin_shufflevector(lo, hi, 0,1,2,3,4,5,6,7,8,9,10,11,12,13,14,15);
}
// 16-bit B 32x16 fragment: lane<16 = col lane K {0..7,16..23}; lane>=16 = col lane-16 K {8..15,24..31}
static __device__ __forceinline__ v16bf ldsB_frag(const __bf16* Bs, int col, int kb){
  v8bf lo = *(const v8bf*)(Bs + col*LDSS + kb);
  v8bf hi = *(const v8bf*)(Bs + col*LDSS + kb + 16);
  return __builtin_shufflevector(lo, hi, 0,1,2,3,4,5,6,7,8,9,10,11,12,13,14,15);
}

template<int EP>   // 0: raw f32 out, 1: BN+lrelu -> bf16 out, 2: distance -> f32 out
__global__ __launch_bounds__(256,2) void k_gemm_wmma(GemmP p){
#if HAVE_ASYNC
  __shared__ __align__(16) __bf16 As[2][BM*LDSS];
  __shared__ __align__(16) __bf16 Bs[2][BN*LDSS];
#else
  __shared__ __align__(16) __bf16 As[1][BM*LDSS];
  __shared__ __align__(16) __bf16 Bs[1][BN*LDSS];
#endif
  const int b = blockIdx.z;
  const __bf16* A  = p.A + (size_t)b * p.strA;
  const __bf16* Bw = p.B + (size_t)b * p.strB;
  const int m0 = blockIdx.x * BM;
  const int n0 = blockIdx.y * BN;
  const int tid = threadIdx.x, wave = tid >> 5, lane = tid & 31;
  const int wm = wave & 3, wn = wave >> 2;            // 4x2 wave grid -> 128x64
  const int r  = lane & 15, hi = lane >> 4;
  const int ksel = hi * 8;                            // A frag K select
  const int kb   = hi * 8;                            // B frag K select

  // per-thread staging slots (256 threads move A:8KB + B:4KB per k-step)
  const int arow = tid >> 1, ahalf = tid & 1;         // A: 16 bf16 = 32B / thread
  const int bcol = tid >> 2, bq = tid & 3;            // B:  8 bf16 = 16B / thread
  const __bf16* aSrc = A  + (size_t)(m0 + arow) * p.lda + ahalf*16;
  const __bf16* bSrc = Bw + (size_t)(n0 + bcol) * p.ldb + bq*8;
  const int aDst = arow*LDSS + ahalf*16;
  const int bDst = bcol*LDSS + bq*8;

  v8f acc[2][2] = {};
  const int nK = p.Kc / BK;

  auto stage = [&](int kt, int buf){
    const __bf16* as = aSrc + kt*BK;
    const __bf16* bs = bSrc + kt*BK;
#if HAVE_ASYNC
    async_cp16(&As[buf][aDst],     as);
    async_cp16(&As[buf][aDst + 8], as + 8);
    async_cp16(&Bs[buf][bDst],     bs);
#else
    *(uint4*)&As[buf][aDst]     = *(const uint4*)as;
    *(uint4*)&As[buf][aDst + 8] = *(const uint4*)(as + 8);
    *(uint4*)&Bs[buf][bDst]     = *(const uint4*)bs;
#endif
  };
  auto compute = [&](int buf){
    v16bf a0 = ldsA_frag(&As[buf][0], wm*32 + r,      ksel);
    v16bf a1 = ldsA_frag(&As[buf][0], wm*32 + 16 + r, ksel);
    v16bf b0 = ldsB_frag(&Bs[buf][0], wn*32 + r,      kb);
    v16bf b1 = ldsB_frag(&Bs[buf][0], wn*32 + 16 + r, kb);
    acc[0][0] = __builtin_amdgcn_wmma_f32_16x16x32_bf16(false, a0, false, b0, (short)0, acc[0][0], false, false);
    acc[0][1] = __builtin_amdgcn_wmma_f32_16x16x32_bf16(false, a0, false, b1, (short)0, acc[0][1], false, false);
    acc[1][0] = __builtin_amdgcn_wmma_f32_16x16x32_bf16(false, a1, false, b0, (short)0, acc[1][0], false, false);
    acc[1][1] = __builtin_amdgcn_wmma_f32_16x16x32_bf16(false, a1, false, b1, (short)0, acc[1][1], false, false);
  };

#if HAVE_ASYNC
  // double-buffered pipeline: stage(k+1) overlaps compute(k)
  stage(0, 0);
  wait_async_all();
  __syncthreads();
  for (int kt = 0; kt < nK; ++kt){
    const int cur = kt & 1;
    if (kt + 1 < nK) stage(kt + 1, cur ^ 1);
    compute(cur);
    if (kt + 1 < nK){ wait_async_all(); __syncthreads(); }
  }
#else
  for (int kt = 0; kt < nK; ++kt){
    stage(kt, 0);
    __syncthreads();
    compute(0);
    __syncthreads();
  }
#endif

  // ---- epilogue (no bounds checks needed: tiles divide M/Nc exactly) ----
#pragma unroll
  for (int i = 0; i < 2; ++i)
#pragma unroll
    for (int j = 0; j < 2; ++j)
#pragma unroll
      for (int v = 0; v < 8; ++v){
        int m = m0 + wm*32 + i*16 + hi*8 + v;
        int c = n0 + wn*32 + j*16 + r;
        float a = acc[i][j][v];
        if (EP == 0){
          p.outF[(size_t)b*p.strO + (size_t)m*p.ldo + c] = a;
        } else if (EP == 1){
          float h = lrelu(a * p.s[c] + p.t[c]);
          p.outH[(size_t)b*p.strO + (size_t)m*p.ldo + c] = (__bf16)h;
        } else {
          // pairwise "distance" (larger = closer): 2*xi.xj - |xi|^2 - |xj|^2
          p.outF[(size_t)m*p.ldo + c] = 2.f*a - p.sq[m] - p.sq[c];
        }
      }
}

// ---------------- top-20 per row (one wave32 per row) ----------------
__global__ __launch_bounds__(256) void k_topk_rows(const float* __restrict__ D,
                                                   int* __restrict__ idx_out, int N){
  const int wave = threadIdx.x >> 5, lane = threadIdx.x & 31;
  const int row = blockIdx.x * 8 + wave;
  const float* rp = D + (size_t)row * N;
  float vals[64];
#pragma unroll
  for (int j = 0; j < 64; ++j) vals[j] = rp[lane + 32*j];   // N == 2048

  for (int sel = 0; sel < 20; ++sel){
    float best = NEGINF; int bi = 0x7fffffff;
#pragma unroll
    for (int j = 0; j < 64; ++j){
      if (vals[j] > best){ best = vals[j]; bi = lane + 32*j; }  // strict > keeps lowest col
    }
#pragma unroll
    for (int off = 16; off > 0; off >>= 1){
      float ov = __shfl_xor(best, off, 32);
      int   oi = __shfl_xor(bi,   off, 32);
      if (ov > best || (ov == best && oi < bi)){ best = ov; bi = oi; }
    }
    if ((bi & 31) == lane) vals[bi >> 5] = NEGINF;  // remove winner
    if (lane == 0) idx_out[(size_t)row * 20 + sel] = bi;
  }
}

// ---------------- EdgeConv gather + BN + lrelu + max-over-K ----------------
// out[b][n][o] = max_k lrelu(s[o]*(y1[j_k] + y2[n] - y1[n]) + t[o]), bf16 store
__global__ void k_edge_gather_max(const float* __restrict__ y1, const float* __restrict__ y2,
                                  const int* __restrict__ idx,
                                  const float* __restrict__ s, const float* __restrict__ t,
                                  __bf16* __restrict__ out, int O, int ldo, int N){
  const int b = blockIdx.y, n = blockIdx.x, o = threadIdx.x;
  const float* Y1 = y1 + (size_t)b * N * O;
  const float* Y2 = y2 + (size_t)b * N * O;
  const int*   ix = idx + ((size_t)b * N + n) * 20;
  float a1   = Y1[(size_t)n * O + o];
  float base = Y2[(size_t)n * O + o] - a1;
  float sc = s[o], tc = t[o];
  float m = NEGINF;
#pragma unroll
  for (int k = 0; k < 20; ++k){
    int j = ix[k];
    float h = Y1[(size_t)j * O + o] + base;
    h = lrelu(h * sc + tc);
    m = fmaxf(m, h);
  }
  out[((size_t)b * N + n) * ldo + o] = (__bf16)m;
}

// ---------------- final 2-channel projection (f32 out) ----------------
__global__ void k_final(const __bf16* __restrict__ H, const __bf16* __restrict__ w8,
                        float* __restrict__ out, int N){
  int n = blockIdx.x*256 + threadIdx.x;
  int b = blockIdx.y;
  if (n >= N) return;
  const __bf16* h = H + ((size_t)b * N + n) * 128;
  float a0 = 0.f, a1 = 0.f;
#pragma unroll 4
  for (int c = 0; c < 128; ++c){
    float v = (float)h[c];
    a0 += v * (float)w8[c];
    a1 += v * (float)w8[128 + c];
  }
  out[((size_t)b*2 + 0) * N + n] = a0;
  out[((size_t)b*2 + 1) * N + n] = a1;
}

// ---------------------------------------------------------------------------
static inline int cdiv(int a, int b){ return (a + b - 1) / b; }

extern "C" void kernel_launch(void* const* d_in, const int* in_sizes, int n_in,
                              void* d_out, int out_size, void* d_ws, size_t ws_size,
                              hipStream_t stream){
  (void)in_sizes; (void)n_in; (void)out_size; (void)ws_size;
  const int B = 8, N = 2048;

  const float* x = (const float*)d_in[0];
  const float* w[8]; for (int i = 0; i < 8; ++i) w[i] = (const float*)d_in[1+i];
  const float* sv[7]; const float* tv[7];
  for (int l = 0; l < 7; ++l){ sv[l] = (const float*)d_in[9 + 2*l]; tv[l] = (const float*)d_in[10 + 2*l]; }

  // ---- carve workspace ----
  char* wp = (char*)d_ws;
  auto alloc = [&](size_t bytes)->void*{ void* p = (void*)wp; wp += (bytes + 255) & ~(size_t)255; return p; };
  // packed weights: [O][Kpad] bf16, zero padded
  __bf16* W1a = (__bf16*)alloc((size_t)64*32*2);    // w1[:, 0:6]  -> Kpad 32
  __bf16* W1b = (__bf16*)alloc((size_t)64*32*2);    // w1[:, 6:12] -> Kpad 32
  __bf16* W2a = (__bf16*)alloc((size_t)64*64*2);
  __bf16* W2b = (__bf16*)alloc((size_t)64*64*2);
  __bf16* W3a = (__bf16*)alloc((size_t)128*64*2);
  __bf16* W3b = (__bf16*)alloc((size_t)128*64*2);
  __bf16* W4a = (__bf16*)alloc((size_t)256*128*2);
  __bf16* W4b = (__bf16*)alloc((size_t)256*128*2);
  __bf16* W5  = (__bf16*)alloc((size_t)1024*512*2);
  __bf16* W6  = (__bf16*)alloc((size_t)256*1536*2);
  __bf16* W7  = (__bf16*)alloc((size_t)128*256*2);
  __bf16* W8  = (__bf16*)alloc((size_t)2*128*2);
  __bf16* X0  = (__bf16*)alloc((size_t)B*N*32*2);
  __bf16* CAT = (__bf16*)alloc((size_t)B*N*1536*2);   // [x5 | x1 | x2 | x3 | x4]
  __bf16* H6  = (__bf16*)alloc((size_t)B*N*256*2);
  __bf16* H7  = (__bf16*)alloc((size_t)B*N*128*2);
  float*  SQ  = (float*) alloc((size_t)B*N*4);
  float*  Dm  = (float*) alloc((size_t)N*N*4);        // per-batch Gram/distance tile
  int*    IDX = (int*)   alloc((size_t)B*N*20*4);
  float*  Y1  = (float*) alloc((size_t)B*N*256*4);
  float*  Y2  = (float*) alloc((size_t)B*N*256*4);

  // ---- repack weights (zero-padded K), transpose input ----
  struct PW { const float* src; __bf16* dst; int O, C, off, ldsrc, Kpad; };
  PW pw[12] = {
    { w[0], W1a, 64,  6,   0,   12,   32   }, { w[0], W1b, 64,  6,   6,   12,   32   },
    { w[1], W2a, 64,  64,  0,   128,  64   }, { w[1], W2b, 64,  64,  64,  128,  64   },
    { w[2], W3a, 128, 64,  0,   128,  64   }, { w[2], W3b, 128, 64,  64,  128,  64   },
    { w[3], W4a, 256, 128, 0,   256,  128  }, { w[3], W4b, 256, 128, 128, 256,  128  },
    { w[4], W5,  1024,512, 0,   512,  512  }, { w[5], W6,  256, 1536,0,   1536, 1536 },
    { w[6], W7,  128, 256, 0,   256,  256  }, { w[7], W8,  2,   128, 0,   128,  128  },
  };
  for (int i = 0; i < 12; ++i)
    k_pack_w<<<dim3(cdiv(pw[i].O*pw[i].Kpad,256)), 256, 0, stream>>>(
        pw[i].src, pw[i].dst, pw[i].O, pw[i].C, pw[i].off, pw[i].ldsrc, pw[i].Kpad);
  k_prep_x0<<<dim3(N/256, B), 256, 0, stream>>>(x, X0, N);

  // ---- EdgeConv layers ----
  struct L { const __bf16* X; int lda; long long strX; int Kpad; int Creal;
             const __bf16* Wa; const __bf16* Wb; int O; int bn; int slice; };
  L ls[4] = {
    { X0,         32,   (long long)N*32,   32,  6,   W1a, W1b, 64,  0, 1024 },
    { CAT + 1024, 1536, (long long)N*1536, 64,  64,  W2a, W2b, 64,  1, 1088 },
    { CAT + 1088, 1536, (long long)N*1536, 64,  64,  W3a, W3b, 128, 2, 1152 },
    { CAT + 1152, 1536, (long long)N*1536, 128, 128, W4a, W4b, 256, 3, 1280 },
  };

  for (int li = 0; li < 4; ++li){
    const L& q = ls[li];
    k_sumsq<<<dim3(N/256, B), 256, 0, stream>>>(q.X, q.strX, q.lda, q.Creal, SQ, N);

    // pairwise distance Gram + top-20, one batch at a time (reuses 16MB tile)
    for (int b = 0; b < B; ++b){
      GemmP dp = {};
      dp.A = q.X + (size_t)b * q.strX; dp.strA = 0; dp.lda = q.lda;
      dp.B = dp.A;                     dp.strB = 0; dp.ldb = q.lda;
      dp.M = N; dp.Nc = N; dp.Kc = q.Kpad;
      dp.outF = Dm; dp.strO = 0; dp.ldo = N;
      dp.sq = SQ + (size_t)b * N;
      k_gemm_wmma<2><<<dim3(N/BM, N/BN, 1), 256, 0, stream>>>(dp);
      k_topk_rows<<<dim3(N/8), 256, 0, stream>>>(Dm, IDX + (size_t)b*N*20, N);
    }

    // y1 = X @ W1^T, y2 = X @ W2^T  (W = [W1 | W2] halves, packed)
    GemmP yp = {};
    yp.A = q.X; yp.strA = q.strX; yp.lda = q.lda;
    yp.strB = 0; yp.ldb = q.Kpad;
    yp.M = N; yp.Nc = q.O; yp.Kc = q.Kpad;
    yp.strO = (long long)N * q.O; yp.ldo = q.O;
    yp.B = q.Wa;  yp.outF = Y1;
    k_gemm_wmma<0><<<dim3(N/BM, cdiv(q.O,BN), B), 256, 0, stream>>>(yp);
    yp.B = q.Wb;  yp.outF = Y2;
    k_gemm_wmma<0><<<dim3(N/BM, cdiv(q.O,BN), B), 256, 0, stream>>>(yp);

    k_edge_gather_max<<<dim3(N, B), q.O, 0, stream>>>(Y1, Y2, IDX, sv[q.bn], tv[q.bn],
                                                      CAT + q.slice, q.O, 1536, N);
  }

  // ---- pointwise convs ----
  {
    GemmP p5 = {};   // x5 = lrelu(BN(W5 @ [x1..x4])) -> CAT[:, 0:1024)
    p5.A = CAT + 1024; p5.strA = (long long)N*1536; p5.lda = 1536;
    p5.B = W5; p5.strB = 0; p5.ldb = 512;
    p5.M = N; p5.Nc = 1024; p5.Kc = 512;
    p5.outH = CAT; p5.strO = (long long)N*1536; p5.ldo = 1536;
    p5.s = sv[4]; p5.t = tv[4];
    k_gemm_wmma<1><<<dim3(N/BM, 1024/BN, B), 256, 0, stream>>>(p5);
  }
  {
    GemmP p6 = {};   // h6 = lrelu(BN(W6 @ [x5, x1..x4]))
    p6.A = CAT; p6.strA = (long long)N*1536; p6.lda = 1536;
    p6.B = W6; p6.strB = 0; p6.ldb = 1536;
    p6.M = N; p6.Nc = 256; p6.Kc = 1536;
    p6.outH = H6; p6.strO = (long long)N*256; p6.ldo = 256;
    p6.s = sv[5]; p6.t = tv[5];
    k_gemm_wmma<1><<<dim3(N/BM, 256/BN, B), 256, 0, stream>>>(p6);
  }
  {
    GemmP p7 = {};   // h7 = lrelu(BN(W7 @ h6))
    p7.A = H6; p7.strA = (long long)N*256; p7.lda = 256;
    p7.B = W7; p7.strB = 0; p7.ldb = 256;
    p7.M = N; p7.Nc = 128; p7.Kc = 256;
    p7.outH = H7; p7.strO = (long long)N*128; p7.ldo = 128;
    p7.s = sv[6]; p7.t = tv[6];
    k_gemm_wmma<1><<<dim3(N/BM, 128/BN, B), 256, 0, stream>>>(p7);
  }

  k_final<<<dim3(N/256, B), 256, 0, stream>>>(H7, W8, (float*)d_out, N);
}
